// PhaseNN_38577396253388
// MI455X (gfx1250) — matrix operand order repaired and checked by reading, not compiled, not run on hardware
//
#include <hip/hip_runtime.h>
#include <math.h>

// ---------------- problem constants ----------------
#define D_IN   784
#define NN     1024          // N oscillators
#define PP     128           // memory patterns
#define CC     10
#define BATCH  256
#define NSTEPS 16
#define KPAD   800           // 784 padded to multiple of 32

#define TWO_PI_F   6.2831853071795864769f
#define T_TOT      0.5f
#define BETA_F     2.0f
#define K_COUP_F   1.0f
#define A_ANC_F    0.08f
#define OMEGA_F    (TWO_PI_F * 200.0f)

typedef _Float16 half_t;
typedef _Float16 v8h  __attribute__((ext_vector_type(8)));
typedef _Float16 v16h __attribute__((ext_vector_type(16)));
typedef float    v8f  __attribute__((ext_vector_type(8)));

// ---------------- WMMA fragment helpers ----------------
// A-matrix 16x32 f16 frag from row-major src (ld in halfs).
// lane l: row = l&15; VGPRs 0-3 hold K = (l>>4)*8 .. +7, VGPRs 4-7 hold K = (l>>4)*8+16 .. +23
__device__ __forceinline__ v16h load_frag_a(const half_t* __restrict__ base, int ld,
                                            int row0, int k0) {
    const int l = threadIdx.x & 31;
    const half_t* p = base + (size_t)(row0 + (l & 15)) * ld + k0 + ((l >> 4) << 3);
    v8h lo = *(const v8h*)(p);
    v8h hi = *(const v8h*)(p + 16);
    return __builtin_shufflevector(lo, hi, 0,1,2,3,4,5,6,7,8,9,10,11,12,13,14,15);
}

// B-matrix 32x16 f16 frag. bt is B^T stored row-major as (Ncols, K) so each B column is
// contiguous in K. lane l: col = l&15, K = k0 + (l>>4)*8 .. +7 and k0 + (l>>4)*8+16 .. +23.
__device__ __forceinline__ v16h load_frag_b(const half_t* __restrict__ bt, int ld,
                                            int k0, int n0) {
    const int l = threadIdx.x & 31;
    const half_t* p = bt + (size_t)(n0 + (l & 15)) * ld + k0 + ((l >> 4) << 3);
    v8h lo = *(const v8h*)(p);
    v8h hi = *(const v8h*)(p + 16);
    return __builtin_shufflevector(lo, hi, 0,1,2,3,4,5,6,7,8,9,10,11,12,13,14,15);
}

__device__ __forceinline__ v8f wmma16(v16h a, v16h b, v8f c) {
    return __builtin_amdgcn_wmma_f32_16x16x32_f16(false, a, false, b, (short)0, c,
                                                  false, false);
}

// ---------------- prep kernels ----------------
// cos/sin of xi in two layouts: Gm (P,2N) row-major (B-side of m-GEMM),
// Gnc/Gns (N,P) N-major (B-side of coupling GEMM).
__global__ void k_prep_xi(const float* __restrict__ xi, half_t* __restrict__ Gm,
                          half_t* __restrict__ Gnc, half_t* __restrict__ Gns) {
    int idx = blockIdx.x * blockDim.x + threadIdx.x;   // P*N threads
    int p = idx >> 10, n = idx & 1023;
    float s, c;
    __sincosf(xi[idx], &s, &c);
    Gm[(size_t)p * (2 * NN) + n]      = (half_t)c;
    Gm[(size_t)p * (2 * NN) + NN + n] = (half_t)s;
    Gnc[(size_t)n * PP + p] = (half_t)c;
    Gns[(size_t)n * PP + p] = (half_t)s;
}

// float -> f16 with K zero-padding (rows x kout), src is (rows x kin) row-major
__global__ void k_pad_half(const float* __restrict__ src, half_t* __restrict__ dst,
                           int kin, int kout) {
    int idx = blockIdx.x * blockDim.x + threadIdx.x;
    int r = idx / kout, k = idx - r * kout;
    dst[idx] = (k < kin) ? (half_t)src[(size_t)r * kin + k] : (half_t)0.0f;
}

// ---------------- encoder: phi0 = 2pi * sigmoid(x W_enc^T + b) ----------------
__global__ void k_encoder(const half_t* __restrict__ Xh, const half_t* __restrict__ Weh,
                          const float* __restrict__ b_enc, float* __restrict__ phiBase,
                          float* __restrict__ phiStage) {
    const int b0 = blockIdx.x * 16;
    const int n0 = blockIdx.y * 128 + (threadIdx.x >> 5) * 16;
    v8f acc = {};
#pragma unroll 5
    for (int k0 = 0; k0 < KPAD; k0 += 32)
        acc = wmma16(load_frag_a(Xh, KPAD, b0, k0), load_frag_b(Weh, KPAD, k0, n0), acc);
    const int l = threadIdx.x & 31, col = l & 15, rb = (l >> 4) << 3;
#pragma unroll
    for (int j = 0; j < 8; ++j) {
        int b = b0 + j + rb, n = n0 + col;
        float z = acc[j] + b_enc[n];
        float phi = TWO_PI_F / (1.0f + expf(-z));
        size_t o = (size_t)b * NN + n;
        phiBase[o] = phi;
        phiStage[o] = phi;
    }
}

// ---------------- stage kernel 1: trig features F = [cos phi, sin phi] (f16) ----
__global__ void k_trig(const float* __restrict__ phi, half_t* __restrict__ F) {
    int idx = blockIdx.x * blockDim.x + threadIdx.x;   // B*N threads
    int b = idx >> 10, n = idx & 1023;
    float s, c;
    __sincosf(phi[idx], &s, &c);
    F[(size_t)b * (2 * NN) + n]      = (half_t)c;
    F[(size_t)b * (2 * NN) + NN + n] = (half_t)s;
}

// ---------------- stage kernel 2: m = F Gm^T (K=2N), then row softmax over P ----
// one block = 16 batch rows, 8 waves each own a 16-wide P tile
__global__ void k_m_softmax(const half_t* __restrict__ F, const half_t* __restrict__ Gm,
                            half_t* __restrict__ wbuf) {
    __shared__ float mtab[16][PP];
    __shared__ float red[16][16];
    __shared__ float rowmax[16], rowsum[16];
    const int b0 = blockIdx.x * 16;
    const int p0 = (threadIdx.x >> 5) * 16;
    v8f acc = {};
#pragma unroll 4
    for (int k0 = 0; k0 < 2 * NN; k0 += 32) {
        __builtin_prefetch(F + (size_t)b0 * (2 * NN) + k0 + 256, 0, 1);
        acc = wmma16(load_frag_a(F, 2 * NN, b0, k0), load_frag_b(Gm, 2 * NN, k0, p0), acc);
    }
    const int l = threadIdx.x & 31, col = l & 15, rb = (l >> 4) << 3;
#pragma unroll
    for (int j = 0; j < 8; ++j) mtab[j + rb][p0 + col] = acc[j];
    __syncthreads();

    const float scale = BETA_F / (float)NN;
    const int row = threadIdx.x >> 4, seg = threadIdx.x & 15;   // 8 p's per thread
    float lmax = -1e30f;
#pragma unroll
    for (int q = 0; q < 8; ++q) lmax = fmaxf(lmax, mtab[row][seg * 8 + q] * scale);
    red[row][seg] = lmax;
    __syncthreads();
    if (seg == 0) {
        float m = red[row][0];
#pragma unroll
        for (int i = 1; i < 16; ++i) m = fmaxf(m, red[row][i]);
        rowmax[row] = m;
    }
    __syncthreads();
    const float rm = rowmax[row];
    float lsum = 0.0f;
#pragma unroll
    for (int q = 0; q < 8; ++q) {
        float e = __expf(mtab[row][seg * 8 + q] * scale - rm);
        mtab[row][seg * 8 + q] = e;
        lsum += e;
    }
    red[row][seg] = lsum;
    __syncthreads();
    if (seg == 0) {
        float s = 0.0f;
#pragma unroll
        for (int i = 0; i < 16; ++i) s += red[row][i];
        rowsum[row] = s;
    }
    __syncthreads();
    const float inv = 1.0f / rowsum[row];
#pragma unroll
    for (int q = 0; q < 8; ++q)
        wbuf[(size_t)(b0 + row) * PP + seg * 8 + q] = (half_t)(mtab[row][seg * 8 + q] * inv);
}

// ---------------- stage kernel 3: coupling GEMMs + fused RK4 epilogue ----------
// Wc = w @ cos(xi), Ws = w @ sin(xi); k = K*(sin phi * Wc - cos phi * Ws) + anchor.
__global__ void k_couple(const half_t* __restrict__ wbuf, const half_t* __restrict__ Gnc,
                         const half_t* __restrict__ Gns, const float* __restrict__ phiCur,
                         float* __restrict__ phiBase, float* __restrict__ accK,
                         float* __restrict__ phiNext, float t_eval, float wk,
                         float cnext, int first, int last, float dt6) {
    const int b0 = blockIdx.x * 16;
    const int n0 = blockIdx.y * 128 + (threadIdx.x >> 5) * 16;
    v8f wc = {}, wsum = {};
#pragma unroll
    for (int p0 = 0; p0 < PP; p0 += 32) {
        v16h a = load_frag_a(wbuf, PP, b0, p0);
        wc   = wmma16(a, load_frag_b(Gnc, PP, p0, n0), wc);
        wsum = wmma16(a, load_frag_b(Gns, PP, p0, n0), wsum);
    }
    const int l = threadIdx.x & 31, col = l & 15, rb = (l >> 4) << 3;
#pragma unroll
    for (int j = 0; j < 8; ++j) {
        int b = b0 + j + rb, n = n0 + col;
        size_t o = (size_t)b * NN + n;
        float phi = phiCur[o];
        float sp, cp;
        __sincosf(phi, &sp, &cp);
        float kv = K_COUP_F * (sp * wc[j] - cp * wsum[j])
                 + A_ANC_F * __sinf(OMEGA_F * t_eval - phi);
        float a_new = first ? (wk * kv) : (accK[o] + wk * kv);
        accK[o] = a_new;
        if (!last) {
            phiNext[o] = phiBase[o] + cnext * kv;
        } else {
            float nb = phiBase[o] + dt6 * a_new;
            phiBase[o] = nb;
            phiNext[o] = nb;
        }
    }
}

// ---------------- readout: [cos,sin](phi_T) @ W_out^T + b_out -------------------
__global__ void k_readout(const float* __restrict__ phiT, const float* __restrict__ Wout,
                          const float* __restrict__ bout, float* __restrict__ out) {
    __shared__ float cs[NN], sn[NN];
    __shared__ float red[16][8];
    const int b = blockIdx.x;
    for (int n = threadIdx.x; n < NN; n += blockDim.x) {
        float s, c;
        __sincosf(phiT[(size_t)b * NN + n], &s, &c);
        cs[n] = c; sn[n] = s;
    }
    __syncthreads();
    const int c_idx = threadIdx.x >> 3, sub = threadIdx.x & 7;
    if (c_idx < CC) {
        float acc = 0.0f;
        const float* wr = Wout + (size_t)c_idx * (2 * NN);
        for (int n = sub * 128; n < sub * 128 + 128; ++n)
            acc += cs[n] * wr[n] + sn[n] * wr[NN + n];
        red[c_idx][sub] = acc;
    }
    __syncthreads();
    if (c_idx < CC && sub == 0) {
        float s = 0.0f;
#pragma unroll
        for (int i = 0; i < 8; ++i) s += red[c_idx][i];
        out[(size_t)b * CC + c_idx] = s + bout[c_idx];
    }
}

// ---------------- host orchestration ----------------
extern "C" void kernel_launch(void* const* d_in, const int* in_sizes, int n_in,
                              void* d_out, int out_size, void* d_ws, size_t ws_size,
                              hipStream_t stream) {
    (void)in_sizes; (void)n_in; (void)out_size; (void)ws_size;
    const float* x     = (const float*)d_in[0];
    const float* W_enc = (const float*)d_in[1];
    const float* b_enc = (const float*)d_in[2];
    const float* xi    = (const float*)d_in[3];
    const float* W_out = (const float*)d_in[4];
    const float* b_out = (const float*)d_in[5];
    float* out = (float*)d_out;

    char* ws = (char*)d_ws;
    auto carve = [&](size_t bytes) {
        char* p = ws;
        ws += (bytes + 255) & ~(size_t)255;
        return p;
    };
    const size_t BN = (size_t)BATCH * NN;
    float*  phiBase = (float*)carve(BN * 4);
    float*  phiA    = (float*)carve(BN * 4);
    float*  phiB    = (float*)carve(BN * 4);
    float*  accK    = (float*)carve(BN * 4);
    half_t* F       = (half_t*)carve((size_t)BATCH * 2 * NN * 2);
    half_t* wbuf    = (half_t*)carve((size_t)BATCH * PP * 2);
    half_t* Gm      = (half_t*)carve((size_t)PP * 2 * NN * 2);
    half_t* Gnc     = (half_t*)carve((size_t)NN * PP * 2);
    half_t* Gns     = (half_t*)carve((size_t)NN * PP * 2);
    half_t* Xh      = (half_t*)carve((size_t)BATCH * KPAD * 2);
    half_t* Weh     = (half_t*)carve((size_t)NN * KPAD * 2);

    // precompute xi trig tables + f16 padded copies of x / W_enc
    k_prep_xi<<<(PP * NN) / 256, 256, 0, stream>>>(xi, Gm, Gnc, Gns);
    k_pad_half<<<(BATCH * KPAD) / 256, 256, 0, stream>>>(x, Xh, D_IN, KPAD);
    k_pad_half<<<(NN * KPAD) / 256, 256, 0, stream>>>(W_enc, Weh, D_IN, KPAD);

    // encoder
    k_encoder<<<dim3(BATCH / 16, NN / 128), 256, 0, stream>>>(Xh, Weh, b_enc,
                                                              phiBase, phiA);

    // RK4 time loop: 16 steps x 4 stages
    const float dt = T_TOT / (float)NSTEPS;
    const float toff[4]  = {0.0f, 0.5f * dt, 0.5f * dt, dt};
    const float wk[4]    = {1.0f, 2.0f, 2.0f, 1.0f};
    const float cnext[4] = {0.5f * dt, 0.5f * dt, dt, 0.0f};
    float* cur = phiA;
    float* nxt = phiB;
    for (int s = 0; s < NSTEPS; ++s) {
        const float ts = (float)s * dt;
        for (int i = 0; i < 4; ++i) {
            k_trig<<<(int)(BN / 256), 256, 0, stream>>>(cur, F);
            k_m_softmax<<<BATCH / 16, 256, 0, stream>>>(F, Gm, wbuf);
            k_couple<<<dim3(BATCH / 16, NN / 128), 256, 0, stream>>>(
                wbuf, Gnc, Gns, cur, phiBase, accK, nxt,
                ts + toff[i], wk[i], cnext[i], (i == 0) ? 1 : 0, (i == 3) ? 1 : 0,
                dt / 6.0f);
            float* t = cur; cur = nxt; nxt = t;
        }
    }

    // readout on phi_T (held in phiBase)
    k_readout<<<BATCH, 128, 0, stream>>>(phiBase, W_out, b_out, out);
}
